// MultiHeadAssociativeMemory_71347996721386
// MI455X (gfx1250) — compile-verified, matrix operand order checked by reading, not run on hardware
//
#include <hip/hip_runtime.h>

// ---------------------------------------------------------------------------
// MultiHeadAssociativeMemory on gfx1250 (MI455X), WMMA bf16 pipeline with
// CDNA5 async global->LDS double buffering.
// B=4, T=2048, V=512, C=512, H=4, D=128, NB2=128
// ---------------------------------------------------------------------------

typedef __attribute__((ext_vector_type(16))) __bf16 bf16x16;
typedef __attribute__((ext_vector_type(8)))  __bf16 bf16x8;
typedef __attribute__((ext_vector_type(8)))  float  f32x8;

static __device__ inline f32x8 wmma_bf16(bf16x16 a, bf16x16 b, f32x8 c) {
  // v_wmma_f32_16x16x32_bf16  (probe-verified 8-arg signature)
  return __builtin_amdgcn_wmma_f32_16x16x32_bf16(false, a, false, b, (short)0, c,
                                                 false, false);
}

union FragU { bf16x16 v; bf16x8 h[2]; };

// Load one 16x32 A-style fragment row chunk. rowBase points at the 32-wide K
// chunk for this lane's row. Per ISA: lane-half 0 holds K[0..7]+K[16..23],
// lane-half 1 holds K[8..15]+K[24..31].
static __device__ inline bf16x16 load_frag(const __bf16* rowBase, int half) {
  FragU f;
  f.h[0] = *(const bf16x8*)(rowBase + half * 8);
  f.h[1] = *(const bf16x8*)(rowBase + 16 + half * 8);
  return f.v;
}

static __device__ inline void zero8(f32x8* a, int n) {
  for (int i = 0; i < n; ++i)
    for (int r = 0; r < 8; ++r) a[i][r] = 0.f;
}

// LDS byte offset of a __shared__ object (generic pointer low 32 bits).
static __device__ inline unsigned lds_off(const void* p) {
  return (unsigned)(unsigned long long)p;
}

// CDNA5 async copy: 16 bytes per lane, global -> LDS, tracked by ASYNCcnt.
static __device__ inline void async_copy16(unsigned lds_addr, const void* gaddr) {
  asm volatile("global_load_async_to_lds_b128 %0, %1, off"
               :: "v"(lds_addr), "v"((unsigned long long)gaddr)
               : "memory");
}

static __device__ inline void wait_async0() {
  asm volatile("s_wait_asynccnt 0x0" ::: "memory");
}

// ---------------------------------------------------------------------------
// Kernel 0: weight prep + x conversion.
//   xb [8192][512] bf16          = x
//   Wt [1536][512] bf16 (N-major): n<512 -> q_w, n<1024 -> k_w, else v_w
//   owb [512][512] bf16          = o_w (row-major == B^T layout for final GEMM)
// ---------------------------------------------------------------------------
__global__ void k_prepare(const float* __restrict__ x, const float* __restrict__ basis,
                          const float* __restrict__ qw, const float* __restrict__ kw,
                          const float* __restrict__ vc, const float* __restrict__ oc,
                          __bf16* __restrict__ xb, __bf16* __restrict__ Wt,
                          __bf16* __restrict__ owb) {
  long i = (long)blockIdx.x * blockDim.x + threadIdx.x;
  const long NX = 8192L * 512;
  if (i < NX) { xb[i] = (__bf16)x[i]; return; }
  i -= NX;
  if (i < 1536L * 512) {
    int n = (int)(i >> 9), kk = (int)(i & 511);
    float wv;
    if (n < 512) {
      wv = qw[kk * 512 + n];
    } else if (n < 1024) {
      wv = kw[kk * 512 + (n - 512)];
    } else {
      int c = n - 1024;
      float s = 0.f;
      for (int j = 0; j < 128; ++j) s += basis[kk * 128 + j] * vc[c * 128 + j];
      wv = s;
    }
    Wt[(long)n * 512 + kk] = (__bf16)wv;
    return;
  }
  i -= 1536L * 512;
  int v = (int)(i >> 9), c = (int)(i & 511);
  float s = 0.f;
  for (int j = 0; j < 128; ++j) s += basis[v * 128 + j] * oc[c * 128 + j];
  owb[(long)v * 512 + c] = (__bf16)s;
}

// ---------------------------------------------------------------------------
// Generic 128xN WMMA GEMM body: OUT[M][N] = A[M][512] @ Bt[N][512]^T.
// A,Bt bf16 row-major (Bt is N-major so B-fragments load contiguously).
// Async double-buffered LDS staging; one barrier per K chunk.
// ---------------------------------------------------------------------------
template <int LDO>
static __device__ inline void gemm512_body(const __bf16* __restrict__ A,
                                           const __bf16* __restrict__ Bt,
                                           float* __restrict__ outF,
                                           __bf16* __restrict__ outB,
                                           float scale,
                                           __bf16* As /*[2][128*32]*/,
                                           __bf16* Bs /*[2][128*32]*/) {
  const int mBase = blockIdx.x * 128;
  const int nBase = blockIdx.y * 128;
  const int lane = threadIdx.x & 31, w = threadIdx.x >> 5;
  const int half = lane >> 4, l16 = lane & 15;
  f32x8 acc[8];
  zero8(acc, 8);
  const int tr = threadIdx.x >> 1;          // 0..127
  const int tc = (threadIdx.x & 1) * 16;    // 0 / 16
  const __bf16* gA = A + (long)(mBase + tr) * 512 + tc;
  const __bf16* gB = Bt + (long)(nBase + tr) * 512 + tc;

  // Preload chunk 0 into buffer 0.
  async_copy16(lds_off(&As[tr * 32 + tc]),     gA);
  async_copy16(lds_off(&As[tr * 32 + tc + 8]), gA + 8);
  async_copy16(lds_off(&Bs[tr * 32 + tc]),     gB);
  async_copy16(lds_off(&Bs[tr * 32 + tc + 8]), gB + 8);
  wait_async0();
  __syncthreads();

  for (int it = 0; it < 16; ++it) {
    const int cur = (it & 1) * (128 * 32);
    const int nxt = ((it + 1) & 1) * (128 * 32);
    if (it + 1 < 16) {
      const __bf16* nA = gA + (it + 1) * 32;
      const __bf16* nB = gB + (it + 1) * 32;
      async_copy16(lds_off(&As[nxt + tr * 32 + tc]),     nA);
      async_copy16(lds_off(&As[nxt + tr * 32 + tc + 8]), nA + 8);
      async_copy16(lds_off(&Bs[nxt + tr * 32 + tc]),     nB);
      async_copy16(lds_off(&Bs[nxt + tr * 32 + tc + 8]), nB + 8);
    }
    bf16x16 af = load_frag(&As[cur + (w * 16 + l16) * 32], half);
#pragma unroll
    for (int nc = 0; nc < 8; ++nc) {
      bf16x16 bfr = load_frag(&Bs[cur + (nc * 16 + l16) * 32], half);
      acc[nc] = wmma_bf16(af, bfr, acc[nc]);
    }
    wait_async0();
    __syncthreads();
  }
#pragma unroll
  for (int nc = 0; nc < 8; ++nc) {
#pragma unroll
    for (int r = 0; r < 8; ++r) {
      int row = mBase + w * 16 + r + 8 * half;
      int col = nBase + nc * 16 + l16;
      if (outF) outF[(long)row * LDO + col] = acc[nc][r] * scale;
      else      outB[(long)row * LDO + col] = (__bf16)(acc[nc][r] * scale);
    }
  }
}

// Kernel 1: QKV projection.  QKV[8192][1536] = xb @ W (fp32 out, for norms).
__global__ __launch_bounds__(256) void k_proj(const __bf16* __restrict__ xb,
                                              const __bf16* __restrict__ Wt,
                                              float* __restrict__ QKV) {
  __shared__ __bf16 As[2 * 128 * 32];
  __shared__ __bf16 Bs[2 * 128 * 32];
  gemm512_body<1536>(xb, Wt, QKV, nullptr, 1.0f, As, Bs);
}

// Kernel 4: out[8192][512] = Rb @ o_w^T * scale (fp32 out).
__global__ __launch_bounds__(256) void k_out(const __bf16* __restrict__ Rb,
                                             const __bf16* __restrict__ owb,
                                             const float* __restrict__ scale,
                                             float* __restrict__ out) {
  __shared__ __bf16 As[2 * 128 * 32];
  __shared__ __bf16 Bs[2 * 128 * 32];
  gemm512_body<512>(Rb, owb, out, nullptr, scale[0], As, Bs);
}

// ---------------------------------------------------------------------------
// Kernel 2: l2-normalize q,k rows over C=512, convert to bf16 per-head layout.
//   qbh/kbh : [B,H,T,D] row-major      vT : [B,H,D,T]  (B^T for retrieval)
// One wave per (b,t) row.
// ---------------------------------------------------------------------------
__global__ __launch_bounds__(256) void k_norm(const float* __restrict__ QKV,
                                              __bf16* __restrict__ qbh,
                                              __bf16* __restrict__ kbh,
                                              __bf16* __restrict__ vT) {
  const int lane = threadIdx.x & 31, w = threadIdx.x >> 5;
  const int row = blockIdx.x * 8 + w;      // 0..8191
  const int b = row >> 11, t = row & 2047;
  const float* base = QKV + (long)row * 1536;
  float qv[16], kv[16];
  float sq = 0.f, sk = 0.f;
#pragma unroll
  for (int i = 0; i < 16; ++i) {
    qv[i] = base[i * 32 + lane];
    kv[i] = base[512 + i * 32 + lane];
    sq += qv[i] * qv[i];
    sk += kv[i] * kv[i];
  }
#pragma unroll
  for (int off = 16; off >= 1; off >>= 1) {
    sq += __shfl_xor(sq, off, 32);
    sk += __shfl_xor(sk, off, 32);
  }
  const float rq = 1.f / fmaxf(sqrtf(sq), 1e-12f);
  const float rk = 1.f / fmaxf(sqrtf(sk), 1e-12f);
#pragma unroll
  for (int i = 0; i < 16; ++i) {
    int c = i * 32 + lane;
    int h = c >> 7, d = c & 127;
    long p = (((long)b * 4 + h) * 2048 + t) * 128 + d;
    qbh[p] = (__bf16)(qv[i] * rq);
    kbh[p] = (__bf16)(kv[i] * rk);
    vT[(((long)b * 4 + h) * 128 + d) * 2048 + t] = (__bf16)base[1024 + c];
  }
}

// ---------------------------------------------------------------------------
// Kernel 3: anti-causal decayed attention (flash style).
// One WG per (qb, h, b). mask: j > i, weight decay^(j-i-1). Loops kb >= qb.
// K/V tiles async double-buffered; scores processed in 32-key chunks so the
// bf16 restage buffer is only 16x32 per wave.
// ---------------------------------------------------------------------------
__global__ __launch_bounds__(256) void k_attn(const __bf16* __restrict__ q,
                                              const __bf16* __restrict__ k,
                                              const __bf16* __restrict__ vT,
                                              const float* __restrict__ decay_logits,
                                              __bf16* __restrict__ Rb) {
  __shared__ __bf16 Ks[2 * 128 * 128];   // key rows [j][d], double buffered
  __shared__ __bf16 Vs[2 * 128 * 128];   // v^T rows [d][j], double buffered
  __shared__ __bf16 Ss[8 * 16 * 32];     // per-wave 16x32 score staging
  const int qb = blockIdx.x, h = blockIdx.y, b = blockIdx.z;
  const int lane = threadIdx.x & 31, w = threadIdx.x >> 5;
  const int half = lane >> 4, l16 = lane & 15;
  const __bf16* qp = q + (((long)b * 4 + h) * 2048) * 128;
  const __bf16* kp = k + (((long)b * 4 + h) * 2048) * 128;
  const __bf16* vp = vT + (((long)b * 4 + h) * 128) * 2048;
  const float logit = decay_logits[h];
  const float dec = 1.f / (1.f + __expf(-logit));
  const float log2d = __log2f(dec);

  bf16x16 qf[4];
#pragma unroll
  for (int dc = 0; dc < 4; ++dc)
    qf[dc] = load_frag(qp + (long)(qb * 128 + w * 16 + l16) * 128 + dc * 32, half);

  f32x8 racc[8];
  zero8(racc, 8);

  const int tr = threadIdx.x >> 1;          // 0..127
  const int tc = (threadIdx.x & 1) * 64;    // 0 / 64
  const int nblocks = 16 - qb;

  // Issue async copies of one 128x128 K tile + V^T tile into buffer `buf`.
  auto issue_tile = [&](int kb, int buf) {
    const int jBase = kb * 128;
    const int o = buf * (128 * 128);
#pragma unroll
    for (int u = 0; u < 8; ++u) {
      async_copy16(lds_off(&Ks[o + tr * 128 + tc + u * 8]),
                   &kp[(long)(jBase + tr) * 128 + tc + u * 8]);
      async_copy16(lds_off(&Vs[o + tr * 128 + tc + u * 8]),
                   &vp[(long)tr * 2048 + jBase + tc + u * 8]);
    }
  };

  issue_tile(qb, 0);
  wait_async0();
  __syncthreads();

  for (int ib = 0; ib < nblocks; ++ib) {
    const int kb = qb + ib;
    const int jBase = kb * 128;
    const int cur = (ib & 1) * (128 * 128);
    if (ib + 1 < nblocks) issue_tile(kb + 1, (ib + 1) & 1);

    __bf16* sw = &Ss[w * 16 * 32];
#pragma unroll
    for (int kc = 0; kc < 4; ++kc) {
      // scores for keys [jBase + kc*32, +32)
      f32x8 sacc[2];
      zero8(sacc, 2);
#pragma unroll
      for (int n2 = 0; n2 < 2; ++n2) {
        const int nc = kc * 2 + n2;
#pragma unroll
        for (int dc = 0; dc < 4; ++dc) {
          bf16x16 kf = load_frag(&Ks[cur + (nc * 16 + l16) * 128 + dc * 32], half);
          sacc[n2] = wmma_bf16(qf[dc], kf, sacc[n2]);
        }
      }
      // decay weight + restage as bf16 A-fragment (16 rows x 32 keys)
#pragma unroll
      for (int n2 = 0; n2 < 2; ++n2) {
#pragma unroll
        for (int r = 0; r < 8; ++r) {
          int i_g = qb * 128 + w * 16 + r + 8 * half;
          int j_g = jBase + (kc * 2 + n2) * 16 + l16;
          float wt = (j_g > i_g) ? exp2f(log2d * (float)(j_g - i_g - 1)) : 0.f;
          sw[(r + 8 * half) * 32 + n2 * 16 + l16] = (__bf16)(sacc[n2][r] * wt);
        }
      }
      // retrieval update for this 32-key chunk
      bf16x16 sf = load_frag(&sw[l16 * 32], half);
#pragma unroll
      for (int dc2 = 0; dc2 < 8; ++dc2) {
        bf16x16 vf = load_frag(&Vs[cur + (dc2 * 16 + l16) * 128 + kc * 32], half);
        racc[dc2] = wmma_bf16(sf, vf, racc[dc2]);
      }
    }
    wait_async0();
    __syncthreads();
  }

  // Store retrieved back to [B,T,C] as bf16 (feeds the output GEMM directly).
#pragma unroll
  for (int dc2 = 0; dc2 < 8; ++dc2) {
#pragma unroll
    for (int r = 0; r < 8; ++r) {
      int row = qb * 128 + w * 16 + r + 8 * half;
      int col = h * 128 + dc2 * 16 + l16;
      Rb[((long)b * 2048 + row) * 512 + col] = (__bf16)racc[dc2][r];
    }
  }
}

// ---------------------------------------------------------------------------
// Launch. Workspace layout (bytes):
//   xb  @ 0         (8 MB)    Wt  @ 8388608  (1.5 MB)  owb @ 9961472 (0.5 MB)
//   QKV @ 10485760  (48 MB)   qbh @ 60817408 (8 MB)    kbh @ 69206016 (8 MB)
//   vT  @ 77594624  (8 MB)    Rb  @ 85983232 (8 MB)    total ~94 MB
// ---------------------------------------------------------------------------
extern "C" void kernel_launch(void* const* d_in, const int* in_sizes, int n_in,
                              void* d_out, int out_size, void* d_ws, size_t ws_size,
                              hipStream_t stream) {
  (void)in_sizes; (void)n_in; (void)out_size; (void)ws_size;
  const float* x     = (const float*)d_in[0];
  const float* basis = (const float*)d_in[1];
  const float* qw    = (const float*)d_in[2];
  const float* kw    = (const float*)d_in[3];
  const float* vc    = (const float*)d_in[4];
  const float* oc    = (const float*)d_in[5];
  const float* dl    = (const float*)d_in[6];
  const float* sc    = (const float*)d_in[7];
  float* out = (float*)d_out;
  char* ws = (char*)d_ws;

  __bf16* xb  = (__bf16*)(ws);
  __bf16* Wt  = (__bf16*)(ws + 8388608);
  __bf16* owb = (__bf16*)(ws + 9961472);
  float*  QKV = (float*)(ws + 10485760);
  __bf16* qbh = (__bf16*)(ws + 60817408);
  __bf16* kbh = (__bf16*)(ws + 69206016);
  __bf16* vT  = (__bf16*)(ws + 77594624);
  __bf16* Rb  = (__bf16*)(ws + 85983232);

  k_prepare<<<20480, 256, 0, stream>>>(x, basis, qw, kw, vc, oc, xb, Wt, owb);
  k_proj<<<dim3(64, 12), 256, 0, stream>>>(xb, Wt, QKV);
  k_norm<<<1024, 256, 0, stream>>>(QKV, qbh, kbh, vT);
  k_attn<<<dim3(16, 4, 4), 256, 0, stream>>>(qbh, kbh, vT, dl, Rb);
  k_out<<<dim3(64, 4), 256, 0, stream>>>(Rb, owb, sc, out);
}
